// SingleTaskGNN_52905407152894
// MI455X (gfx1250) — matrix operand order repaired and checked by reading, not compiled, not run on hardware
//
#include <hip/hip_runtime.h>

// ===========================================================================
// AttentiveFP (SingleTaskGNN) forward — MI455X / gfx1250 (wave32, WMMA).
//
//   N=100000 nodes, E=400000 edges, G=4096 graphs, H=256, OUT=512.
//
// All dense GEMMs run on v_wmma_f32_16x16x32_bf16 (bf16 in, fp32 accumulate):
//  - generic 16x64-per-wave GEMM: B fragments batch-loaded per K-step so the
//    scheduler emits one load clause + one wait + 4 back-to-back WMMAs;
//    next-K A line prefetched via global_prefetch_b8
//  - fused GRU cell: x@Wih and h@Whh share WMMA accumulators for the r/z
//    gates; WMMAs interleaved across accumulators to break RAW chains;
//    768 WMMAs per 16-row tile; no [N,768] intermediates in HBM
//  - fused GATEConv edge kernel: gathered concat(x[src],edge_attr) tile staged
//    in LDS, WMMA, hj.att_l reduced in LDS -> [E,256] hj never stored
// Segment softmax: atomicMax on monotonic uint float-keys + atomicAdd sums.
// ===========================================================================

#define ACT_NONE  0
#define ACT_RELU  1
#define ACT_LEAKY 2
#define ACT_ELU   3

#define LRELU_SLOPE 0.01f

typedef __bf16 bf16_t;
typedef bf16_t v16bf __attribute__((ext_vector_type(16)));
typedef float  v8f   __attribute__((ext_vector_type(8)));

struct __align__(16) U4 { unsigned a, b, c, d; };

union FragU { v16bf v; U4 q[2]; unsigned u[8]; };

__device__ __forceinline__ v8f vzero8() {
  v8f z;
#pragma unroll
  for (int i = 0; i < 8; ++i) z[i] = 0.f;
  return z;
}

__device__ __forceinline__ unsigned short f2bf(float f) {
  unsigned u = __float_as_uint(f);
  unsigned r = u + 0x7FFFu + ((u >> 16) & 1u);   // round-to-nearest-even
  return (unsigned short)(r >> 16);
}

// Order-preserving float -> uint key (for atomicMax-based segment max).
__device__ __forceinline__ unsigned fkey(float f) {
  unsigned u = __float_as_uint(f);
  return (u & 0x80000000u) ? ~u : (u | 0x80000000u);
}
__device__ __forceinline__ float fdec(unsigned k) {
  unsigned u = (k & 0x80000000u) ? (k & 0x7FFFFFFFu) : ~k;
  return __uint_as_float(u);
}

__device__ __forceinline__ float act_apply(float v, int act) {
  switch (act) {
    case ACT_RELU:  return v > 0.f ? v : 0.f;
    case ACT_LEAKY: return v > 0.f ? v : LRELU_SLOPE * v;
    case ACT_ELU:   return v > 0.f ? v : (__expf(v) - 1.f);
    default:        return v;
  }
}

// ---------------------------------------------------------------------------
// WMMA fragment plumbing (per CDNA5 ISA 16-bit A / B / 32-bit C-D layouts)
// ---------------------------------------------------------------------------
__device__ __forceinline__ v8f wmma_bf16(v16bf a, v16bf b, v8f c) {
  // (neg_a, A, neg_b, B, c_mod, C, reuse_a, reuse_b)
  return __builtin_amdgcn_wmma_f32_16x16x32_bf16(false, a, false, b, (short)0, c,
                                                 false, false);
}

// A 16x32 bf16, row-major in memory: lane l<16 holds row l, K {k0..k0+7, k0+16..k0+23};
// lane l+16 holds row l, K {k0+8..k0+15, k0+24..k0+31}.
__device__ __forceinline__ v16bf load_fragA(const unsigned short* A, int lda,
                                            int row0, int k0, int lane) {
  int m  = row0 + (lane & 15);
  int kb = k0 + ((lane >> 4) << 3);
  const unsigned short* p = A + (size_t)m * lda + kb;
  FragU f;
  f.q[0] = *(const U4*)(p);
  f.q[1] = *(const U4*)(p + 16);
  return f.v;
}

__device__ __forceinline__ v16bf load_fragA_lds(const unsigned short* As, int lda,
                                                int k0, int lane) {
  int m  = lane & 15;
  int kb = k0 + ((lane >> 4) << 3);
  const unsigned* p0 = (const unsigned*)(As + m * lda + kb);
  const unsigned* p1 = (const unsigned*)(As + m * lda + kb + 16);
  FragU f;
#pragma unroll
  for (int j = 0; j < 4; ++j) { f.u[j] = p0[j]; f.u[4 + j] = p1[j]; }
  return f.v;
}

// B 32x16 bf16 where B = W^T and W is [Nc,K] row-major (K contiguous):
// lane n<16 holds column n = W row (n0+n), K {k0..k0+7, k0+16..k0+23}; lane n+16 K {k0+8..k0+15, k0+24..k0+31}.
// -> two 16-byte loads from W.
__device__ __forceinline__ v16bf load_fragB(const unsigned short* W, int ldw,
                                            int n0, int k0, int lane) {
  int n  = n0 + (lane & 15);
  int kb = k0 + ((lane >> 4) << 3);
  const unsigned short* p = W + (size_t)n * ldw + kb;
  FragU f;
  f.q[0] = *(const U4*)(p);
  f.q[1] = *(const U4*)(p + 16);
  return f.v;
}

// ---------------------------------------------------------------------------
// Elementwise / utility kernels
// ---------------------------------------------------------------------------
__global__ void k_fill_u32(unsigned* __restrict__ p, unsigned v, long n) {
  long i = blockIdx.x * (long)blockDim.x + threadIdx.x;
  long s = (long)gridDim.x * blockDim.x;
  for (; i < n; i += s) p[i] = v;
}

// f32 [rows,K] -> bf16 [rows,KP], zero-padding columns K..KP-1.
__global__ void k_cvt_pad(const float* __restrict__ in, unsigned short* __restrict__ out,
                          long rows, int K, int KP) {
  long total = rows * (long)KP;
  long i = blockIdx.x * (long)blockDim.x + threadIdx.x;
  long s = (long)gridDim.x * blockDim.x;
  for (; i < total; i += s) {
    long r = i / KP; int c = (int)(i - r * (long)KP);
    float v = (c < K) ? in[r * (long)K + c] : 0.f;
    out[i] = f2bf(v);
  }
}

// X[i] = act(X[i] + bias[i%C]); optionally also emit bf16 copy.
__global__ void k_bias_act(float* __restrict__ X, const float* __restrict__ bias,
                           unsigned short* __restrict__ Ob, int C, long total, int act) {
  long i = blockIdx.x * (long)blockDim.x + threadIdx.x;
  long s = (long)gridDim.x * blockDim.x;
  for (; i < total; i += s) {
    float v = X[i] + (bias ? bias[(int)(i % C)] : 0.f);
    v = act_apply(v, act);
    X[i] = v;
    if (Ob) Ob[i] = f2bf(v);
  }
}

// ---------------------------------------------------------------------------
// Generic WMMA GEMM:  Of[M,Nc] = act(A[M,K](bf16) @ W[Nc,K]^T + bias)
// One wave computes a 16x64 tile (4 N-subtiles).  M%16==0, K%32==0, Nc%64==0.
// B fragments are batch-loaded so loads clause together ahead of the WMMAs.
// ---------------------------------------------------------------------------
__global__ __launch_bounds__(128, 1)
void k_gemm(const unsigned short* __restrict__ A, int K,
            const unsigned short* __restrict__ W,
            const float* __restrict__ bias,
            float* __restrict__ Of, unsigned short* __restrict__ Ob,
            int M, int Nc, int act) {
  int wid  = (blockIdx.x * blockDim.x + threadIdx.x) >> 5;
  int lane = threadIdx.x & 31;
  int nWaveCols = Nc >> 6;
  int total = (M >> 4) * nWaveCols;
  if (wid >= total) return;
  int mt = wid / nWaveCols;
  int n0 = (wid - mt * nWaveCols) << 6;

  v8f acc[4];
#pragma unroll
  for (int t = 0; t < 4; ++t) acc[t] = vzero8();

  // per-lane A base for prefetch
  const unsigned short* aRow =
      A + (size_t)((mt << 4) + (lane & 15)) * K + ((lane >> 4) << 3);

  for (int k0 = 0; k0 < K; k0 += 32) {
    if (k0 + 32 < K) __builtin_prefetch(aRow + k0 + 32, 0, 3);
    v16bf a = load_fragA(A, K, mt << 4, k0, lane);
    v16bf b[4];
#pragma unroll
    for (int t = 0; t < 4; ++t) b[t] = load_fragB(W, K, n0 + (t << 4), k0, lane);
#pragma unroll
    for (int t = 0; t < 4; ++t) acc[t] = wmma_bf16(a, b[t], acc[t]);
  }

  int col = lane & 15, rbase = (lane >> 4) << 3;
#pragma unroll
  for (int t = 0; t < 4; ++t) {
    int n = n0 + (t << 4) + col;
    float bz = bias ? bias[n] : 0.f;
#pragma unroll
    for (int i = 0; i < 8; ++i) {
      int m = (mt << 4) + rbase + i;
      float v = act_apply(acc[t][i] + bz, act);
      size_t o = (size_t)m * Nc + n;
      Of[o] = v;
      if (Ob) Ob[o] = f2bf(v);
    }
  }
}

// ---------------------------------------------------------------------------
// Fused GRU cell + ReLU:  out = relu(gru_cell(xarg, harg))
//   gi = xarg @ Wih^T + bih ; gh = harg @ Whh^T + bhh
//   r = sig(gi_r+gh_r); z = sig(gi_z+gh_z); n = tanh(gi_n + r*gh_n)
//   out = (1-z)*n + z*harg
// Block = 4 waves = one 16-row tile; wave w owns output cols [w*64,w*64+64).
// r/z gates: gi and gh GEMMs share one accumulator; WMMAs interleaved across
// accumulators (S0,S1,S0,S1 then N,H) so same-acc WMMAs are not adjacent.
// ---------------------------------------------------------------------------
__global__ __launch_bounds__(128, 1)
void k_gru(const unsigned short* __restrict__ Xa,   // x-arg  bf16 [M,256]
           const unsigned short* __restrict__ Ha,   // h-arg  bf16 [M,256]
           const float* __restrict__ Hf,            // h-arg  f32  [M,256]
           const unsigned short* __restrict__ Wih,  // [768,256] bf16
           const unsigned short* __restrict__ Whh,  // [768,256] bf16
           const float* __restrict__ bih, const float* __restrict__ bhh,
           float* __restrict__ Of, unsigned short* __restrict__ Ob, int M) {
  int mt   = blockIdx.x;
  int w    = threadIdx.x >> 5;
  int lane = threadIdx.x & 31;
  int n0   = w << 6;

  v8f aS0[4], aS1[4], aN[4], aH[4];
#pragma unroll
  for (int t = 0; t < 4; ++t) { aS0[t] = vzero8(); aS1[t] = vzero8(); aN[t] = vzero8(); aH[t] = vzero8(); }

  for (int k0 = 0; k0 < 256; k0 += 32) {
    v16bf ax = load_fragA(Xa, 256, mt << 4, k0, lane);
    v16bf ah = load_fragA(Ha, 256, mt << 4, k0, lane);
#pragma unroll
    for (int t = 0; t < 4; ++t) {
      int n = n0 + (t << 4);
      // r & z gates: batch 4 B loads, interleave 4 WMMAs across 2 accumulators
      v16bf b0 = load_fragB(Wih, 256, n,       k0, lane);
      v16bf b1 = load_fragB(Whh, 256, n,       k0, lane);
      v16bf b2 = load_fragB(Wih, 256, n + 256, k0, lane);
      v16bf b3 = load_fragB(Whh, 256, n + 256, k0, lane);
      aS0[t] = wmma_bf16(ax, b0, aS0[t]);
      aS1[t] = wmma_bf16(ax, b2, aS1[t]);
      aS0[t] = wmma_bf16(ah, b1, aS0[t]);
      aS1[t] = wmma_bf16(ah, b3, aS1[t]);
      // n gate: two independent accumulators
      v16bf b4 = load_fragB(Wih, 256, n + 512, k0, lane);
      v16bf b5 = load_fragB(Whh, 256, n + 512, k0, lane);
      aN[t] = wmma_bf16(ax, b4, aN[t]);
      aH[t] = wmma_bf16(ah, b5, aH[t]);
    }
  }

  int col = lane & 15, rbase = (lane >> 4) << 3;
#pragma unroll
  for (int t = 0; t < 4; ++t) {
    int n = n0 + (t << 4) + col;
    float br  = bih[n] + bhh[n];
    float bz  = bih[n + 256] + bhh[n + 256];
    float bni = bih[n + 512];
    float bnh = bhh[n + 512];
#pragma unroll
    for (int i = 0; i < 8; ++i) {
      int m = (mt << 4) + rbase + i;
      float r  = 1.f / (1.f + __expf(-(aS0[t][i] + br)));
      float z  = 1.f / (1.f + __expf(-(aS1[t][i] + bz)));
      float nn = tanhf((aN[t][i] + bni) + r * (aH[t][i] + bnh));
      float hp = Hf[(size_t)m * 256 + n];
      float v  = (1.f - z) * nn + z * hp;
      v = v > 0.f ? v : 0.f;                     // reference always wraps in relu
      size_t o = (size_t)m * 256 + n;
      Of[o] = v;
      Ob[o] = f2bf(v);
    }
  }
}

// ---------------------------------------------------------------------------
// Fused GATEConv edge kernel.  One block = one 16-edge tile.
//   hj    = leaky( concat(x[src], edge_attr) @ W1^T )        (never stored)
//   logit = leaky( hj . att_l + xr[dst] )  ->  alpha[e], atomicMax key[dst]
// A tile (16 x 288, zero-padded K) is gathered into LDS.
// ---------------------------------------------------------------------------
__global__ __launch_bounds__(128, 1)
void k_gate(const unsigned short* __restrict__ Xb,   // x bf16 [N,256]
            const float* __restrict__ EA,            // edge_attr f32 [E,10]
            const int* __restrict__ src, const int* __restrict__ dst,
            const unsigned short* __restrict__ W1,   // [256,288] bf16 (padded)
            const float* __restrict__ attL,          // [256]
            const float* __restrict__ xr,            // [N]  = x . att_r
            float* __restrict__ alpha, unsigned* __restrict__ amaxkey) {
  __shared__ __align__(16) unsigned short As[16 * 288];
  __shared__ float edot[16];

  int e0  = blockIdx.x << 4;
  int tid = threadIdx.x;
  if (tid < 16) edot[tid] = 0.f;

  // Stage gathered A tile: 16 rows x 144 dwords.
  for (int d = tid; d < 16 * 144; d += 128) {
    int row = d / 144, dd = d - row * 144, c2 = dd * 2;
    int e = e0 + row;
    unsigned val;
    if (c2 < 256) {
      val = *(const unsigned*)(Xb + (size_t)src[e] * 256 + c2);
    } else if (c2 < 266) {
      float f0 = EA[(size_t)e * 10 + (c2 - 256)];
      float f1 = EA[(size_t)e * 10 + (c2 - 256) + 1];
      val = (unsigned)f2bf(f0) | ((unsigned)f2bf(f1) << 16);
    } else {
      val = 0u;                                   // K padding 266..287
    }
    ((unsigned*)As)[row * 144 + dd] = val;
  }
  __syncthreads();

  int w = tid >> 5, lane = tid & 31;
  int n0 = w << 6;
  v8f acc[4];
#pragma unroll
  for (int t = 0; t < 4; ++t) acc[t] = vzero8();

  for (int k0 = 0; k0 < 288; k0 += 32) {
    v16bf a = load_fragA_lds(As, 288, k0, lane);
    v16bf b[4];
#pragma unroll
    for (int t = 0; t < 4; ++t) b[t] = load_fragB(W1, 288, n0 + (t << 4), k0, lane);
#pragma unroll
    for (int t = 0; t < 4; ++t) acc[t] = wmma_bf16(a, b[t], acc[t]);
  }

  int col = lane & 15, rbase = (lane >> 4) << 3;
  float part[8];
#pragma unroll
  for (int i = 0; i < 8; ++i) part[i] = 0.f;
#pragma unroll
  for (int t = 0; t < 4; ++t) {
    int n = n0 + (t << 4) + col;
    float al = attL[n];
#pragma unroll
    for (int i = 0; i < 8; ++i) {
      float v = acc[t][i];
      v = v > 0.f ? v : LRELU_SLOPE * v;          // leaky on hj
      part[i] += v * al;
    }
  }
#pragma unroll
  for (int i = 0; i < 8; ++i) atomicAdd(&edot[rbase + i], part[i]);
  __syncthreads();

  if (tid < 16) {
    int e = e0 + tid;
    int d = dst[e];
    float lg = edot[tid] + xr[d];
    lg = lg > 0.f ? lg : LRELU_SLOPE * lg;        // leaky on logit
    alpha[e] = lg;
    atomicMax(&amaxkey[d], fkey(lg));
  }
}

// ---------------------------------------------------------------------------
// Per-row dot products (wave per row): oa = X.va, ob = X.vb (vb optional)
// ---------------------------------------------------------------------------
__global__ void k_rowdot(const float* __restrict__ X, int C,
                         const float* __restrict__ va, const float* __restrict__ vb,
                         float* __restrict__ oa, float* __restrict__ ob, int M) {
  int wid  = (blockIdx.x * blockDim.x + threadIdx.x) >> 5;
  int lane = threadIdx.x & 31;
  if (wid >= M) return;
  const float* row = X + (size_t)wid * C;
  float sa = 0.f, sb = 0.f;
  for (int c = lane; c < C; c += 32) {
    float x = row[c];
    sa += x * va[c];
    if (vb) sb += x * vb[c];
  }
#pragma unroll
  for (int off = 16; off > 0; off >>= 1) {
    sa += __shfl_down(sa, off, 32);
    if (vb) sb += __shfl_down(sb, off, 32);
  }
  if (lane == 0) { oa[wid] = sa; if (ob) ob[wid] = sb; }
}

// GAT logits: alpha[e] = leaky(as[srcIdx]+ad[dst[e]]); atomicMax key[dst].
// src==nullptr means identity indices (mol readout: node e -> graph batch[e]).
__global__ void k_edge_logit(const float* __restrict__ as_, const float* __restrict__ ad,
                             const int* __restrict__ src, const int* __restrict__ dst,
                             float* __restrict__ alpha, unsigned* __restrict__ amaxkey,
                             long nE) {
  long i = blockIdx.x * (long)blockDim.x + threadIdx.x;
  long st = (long)gridDim.x * blockDim.x;
  for (; i < nE; i += st) {
    int s = src ? src[i] : (int)i;
    int d = dst[i];
    float lg = as_[s] + ad[d];
    lg = lg > 0.f ? lg : LRELU_SLOPE * lg;
    alpha[i] = lg;
    atomicMax(&amaxkey[d], fkey(lg));
  }
}

__global__ void k_expsum(const float* __restrict__ alpha, const unsigned* __restrict__ amaxkey,
                         const int* __restrict__ dst, float* __restrict__ eb,
                         float* __restrict__ ssum, long nE) {
  long i = blockIdx.x * (long)blockDim.x + threadIdx.x;
  long st = (long)gridDim.x * blockDim.x;
  for (; i < nE; i += st) {
    int d = dst[i];
    float ex = __expf(alpha[i] - fdec(amaxkey[d]));
    eb[i] = ex;
    atomicAdd(&ssum[d], ex);
  }
}

// Acc[dst[e]] += Xm[srcIdx] * (eb[e] / (ssum[dst[e]]+1e-16)); 4 cols/thread.
__global__ void k_scatter_msg(const float* __restrict__ eb, const float* __restrict__ ssum,
                              const int* __restrict__ src, const int* __restrict__ dst,
                              const float* __restrict__ Xm, float* __restrict__ Acc,
                              long nE) {
  long total = nE << 6;                       // H/4 = 64 chunks per edge
  long i = blockIdx.x * (long)blockDim.x + threadIdx.x;
  long st = (long)gridDim.x * blockDim.x;
  for (; i < total; i += st) {
    long e = i >> 6;
    int c = (int)(i & 63) << 2;
    int s = src ? src[e] : (int)e;
    int d = dst[e];
    float a = eb[e] / (ssum[d] + 1e-16f);
    const float4 v = *(const float4*)(Xm + (size_t)s * 256 + c);
    float* o = Acc + (size_t)d * 256 + c;
    atomicAdd(o + 0, v.x * a);
    atomicAdd(o + 1, v.y * a);
    atomicAdd(o + 2, v.z * a);
    atomicAdd(o + 3, v.w * a);
  }
}

// Acc[idx[n]] += X[n]  (graph readout segment-sum)
__global__ void k_scatter_rows(const float* __restrict__ X, const int* __restrict__ idx,
                               float* __restrict__ Acc, long nRows) {
  long total = nRows << 6;
  long i = blockIdx.x * (long)blockDim.x + threadIdx.x;
  long st = (long)gridDim.x * blockDim.x;
  for (; i < total; i += st) {
    long n = i >> 6;
    int c = (int)(i & 63) << 2;
    int g = idx[n];
    const float4 v = *(const float4*)(X + (size_t)n * 256 + c);
    float* o = Acc + (size_t)g * 256 + c;
    atomicAdd(o + 0, v.x);
    atomicAdd(o + 1, v.y);
    atomicAdd(o + 2, v.z);
    atomicAdd(o + 3, v.w);
  }
}

// Final head: out[g] = H1[g,:64] . W2 + b2
__global__ void k_head2(const float* __restrict__ H1, const float* __restrict__ W2,
                        const float* __restrict__ b2, float* __restrict__ out, int Gn) {
  int g = blockIdx.x * blockDim.x + threadIdx.x;
  if (g >= Gn) return;
  float s = b2[0];
#pragma unroll 8
  for (int i = 0; i < 64; ++i) s += H1[(size_t)g * 64 + i] * W2[i];
  out[g] = s;
}

// ---------------------------------------------------------------------------
// Host side
// ---------------------------------------------------------------------------
static inline int gs_blocks(long total, int tpb) {
  long b = (total + tpb - 1) / tpb;
  if (b < 1) b = 1;
  if (b > (1L << 20)) b = (1L << 20);
  return (int)b;
}

static void launch_gemm(hipStream_t s, const unsigned short* A, int K,
                        const unsigned short* W, const float* bias,
                        float* Of, unsigned short* Ob, int M, int Nc, int act) {
  int waves  = (M >> 4) * (Nc >> 6);
  int blocks = (waves + 3) >> 2;
  k_gemm<<<blocks, 128, 0, s>>>(A, K, W, bias, Of, Ob, M, Nc, act);
}

extern "C" void kernel_launch(void* const* d_in, const int* in_sizes, int n_in,
                              void* d_out, int out_size, void* d_ws, size_t ws_size,
                              hipStream_t stream) {
  (void)in_sizes; (void)n_in; (void)out_size; (void)ws_size;
  const int  N = 100000, E = 400000, Gn = 4096;

  // ---- inputs (setup_inputs order; params flattened in JAX tree order =
  //      recursively sorted dict keys) ----
  const float* x_in  = (const float*)d_in[0];
  const int*   eidx  = (const int*)d_in[1];
  const float* eattr = (const float*)d_in[2];
  const int*   batch = (const int*)d_in[3];
  const int* src = eidx;        // edge_index[0]
  const int* dst = eidx + E;    // edge_index[1]

  int q = 4;
  const float *ac_W[2], *ac_adst[2], *ac_asrc[2], *ac_b[2];
  for (int l = 0; l < 2; ++l) {                       // atom_convs: W, att_dst, att_src, bias
    ac_W[l]    = (const float*)d_in[q++];
    ac_adst[l] = (const float*)d_in[q++];
    ac_asrc[l] = (const float*)d_in[q++];
    ac_b[l]    = (const float*)d_in[q++];
  }
  const float *ag_Wih[2], *ag_Whh[2], *ag_bhh[2], *ag_bih[2];
  for (int l = 0; l < 2; ++l) {                       // atom_grus: Wih, Whh, bhh, bih
    ag_Wih[l] = (const float*)d_in[q++];
    ag_Whh[l] = (const float*)d_in[q++];
    ag_bhh[l] = (const float*)d_in[q++];
    ag_bih[l] = (const float*)d_in[q++];
  }
  const float* g_attl = (const float*)d_in[q++];      // gate: att_l, att_r, bias, lin1_W, lin2_W
  const float* g_attr = (const float*)d_in[q++];
  const float* g_bias = (const float*)d_in[q++];
  const float* g_W1   = (const float*)d_in[q++];
  const float* g_W2   = (const float*)d_in[q++];
  const float* g0_Wih = (const float*)d_in[q++];      // gru0: Wih, Whh, bhh, bih
  const float* g0_Whh = (const float*)d_in[q++];
  const float* g0_bhh = (const float*)d_in[q++];
  const float* g0_bih = (const float*)d_in[q++];
  const float* head_W1 = (const float*)d_in[q++];     // head_W1, head_W2, head_b1, head_b2
  const float* head_W2 = (const float*)d_in[q++];
  const float* head_b1 = (const float*)d_in[q++];
  const float* head_b2 = (const float*)d_in[q++];
  const float* lin1_W = (const float*)d_in[q++];      // lin1_W, lin1_b, lin2_W, lin2_b
  const float* lin1_b = (const float*)d_in[q++];
  const float* lin2_W = (const float*)d_in[q++];
  const float* lin2_b = (const float*)d_in[q++];
  const float* mc_W    = (const float*)d_in[q++];     // mol_conv: W, att_dst, att_src, bias
  const float* mc_adst = (const float*)d_in[q++];
  const float* mc_asrc = (const float*)d_in[q++];
  const float* mc_b    = (const float*)d_in[q++];
  const float* mg_Wih = (const float*)d_in[q++];      // mol_gru: Wih, Whh, bhh, bih
  const float* mg_Whh = (const float*)d_in[q++];
  const float* mg_bhh = (const float*)d_in[q++];
  const float* mg_bih = (const float*)d_in[q++];

  // ---- workspace bump allocator ----
  char* wp = (char*)d_ws;
  auto alloc = [&](size_t bytes) -> void* {
    void* r = (void*)wp;
    wp += (bytes + 255) & ~(size_t)255;
    return r;
  };
  unsigned short* xinp = (unsigned short*)alloc((size_t)N * 64 * 2);
  unsigned short* xA_b = (unsigned short*)alloc((size_t)N * 256 * 2);
  unsigned short* xB_b = (unsigned short*)alloc((size_t)N * 256 * 2);
  float*          xA_f = (float*)alloc((size_t)N * 256 * 4);
  float*          xB_f = (float*)alloc((size_t)N * 256 * 4);
  unsigned short* h_b  = (unsigned short*)alloc((size_t)N * 256 * 2);
  float*          h_f  = (float*)alloc((size_t)N * 256 * 4);   // msg accum / elu out
  float*          hx_f = (float*)alloc((size_t)N * 256 * 4);   // lin-transformed nodes
  float*          hd_f = (float*)alloc((size_t)Gn * 256 * 4);
  float*          oA_f = (float*)alloc((size_t)Gn * 256 * 4);
  float*          oB_f = (float*)alloc((size_t)Gn * 256 * 4);
  unsigned short* oA_b = (unsigned short*)alloc((size_t)Gn * 256 * 2);
  unsigned short* oB_b = (unsigned short*)alloc((size_t)Gn * 256 * 2);
  float*          mh_f = (float*)alloc((size_t)Gn * 256 * 4);
  unsigned short* mh_b = (unsigned short*)alloc((size_t)Gn * 256 * 2);
  float*          emb_f = (float*)alloc((size_t)Gn * 512 * 4);
  unsigned short* emb_b = (unsigned short*)alloc((size_t)Gn * 512 * 2);
  float*          h1_f  = (float*)alloc((size_t)Gn * 64 * 4);
  float* xr    = (float*)alloc((size_t)N * 4);
  float* asN   = (float*)alloc((size_t)N * 4);
  float* adN   = (float*)alloc((size_t)N * 4);
  float* alpha = (float*)alloc((size_t)E * 4);
  float* ebuf  = (float*)alloc((size_t)E * 4);
  float* ssum  = (float*)alloc((size_t)N * 4);
  unsigned* akey = (unsigned*)alloc((size_t)N * 4);
  float* adG   = (float*)alloc((size_t)Gn * 4);
  float* ssg   = (float*)alloc((size_t)Gn * 4);
  unsigned* akg = (unsigned*)alloc((size_t)Gn * 4);
  // bf16 weights
  unsigned short* w_lin1 = (unsigned short*)alloc((size_t)256 * 64 * 2);
  unsigned short* w_g1   = (unsigned short*)alloc((size_t)256 * 288 * 2);
  unsigned short* w_g2   = (unsigned short*)alloc((size_t)256 * 256 * 2);
  unsigned short* w_g0ih = (unsigned short*)alloc((size_t)768 * 256 * 2);
  unsigned short* w_g0hh = (unsigned short*)alloc((size_t)768 * 256 * 2);
  unsigned short* w_aW[2], *w_aih[2], *w_ahh[2];
  for (int l = 0; l < 2; ++l) {
    w_aW[l]  = (unsigned short*)alloc((size_t)256 * 256 * 2);
    w_aih[l] = (unsigned short*)alloc((size_t)768 * 256 * 2);
    w_ahh[l] = (unsigned short*)alloc((size_t)768 * 256 * 2);
  }
  unsigned short* w_mW  = (unsigned short*)alloc((size_t)256 * 256 * 2);
  unsigned short* w_mih = (unsigned short*)alloc((size_t)768 * 256 * 2);
  unsigned short* w_mhh = (unsigned short*)alloc((size_t)768 * 256 * 2);
  unsigned short* w_l2  = (unsigned short*)alloc((size_t)512 * 256 * 2);
  unsigned short* w_h1  = (unsigned short*)alloc((size_t)64 * 512 * 2);

#define CVT(inp, outp, R, Kk, KP) \
  k_cvt_pad<<<gs_blocks((long)(R) * (KP), 256), 256, 0, stream>>>((inp), (outp), (R), (Kk), (KP))
#define FILL(p, v, n) \
  k_fill_u32<<<gs_blocks((long)(n), 256), 256, 0, stream>>>((unsigned*)(p), (v), (long)(n))

  // ---- weight conversion (f32 -> bf16, with K padding where needed) ----
  CVT(lin1_W, w_lin1, 256, 39, 64);
  CVT(g_W1,   w_g1,   256, 266, 288);
  CVT(g_W2,   w_g2,   256, 256, 256);
  CVT(g0_Wih, w_g0ih, 768, 256, 256);
  CVT(g0_Whh, w_g0hh, 768, 256, 256);
  for (int l = 0; l < 2; ++l) {
    CVT(ac_W[l],   w_aW[l],  256, 256, 256);
    CVT(ag_Wih[l], w_aih[l], 768, 256, 256);
    CVT(ag_Whh[l], w_ahh[l], 768, 256, 256);
  }
  CVT(mc_W,   w_mW,  256, 256, 256);
  CVT(mg_Wih, w_mih, 768, 256, 256);
  CVT(mg_Whh, w_mhh, 768, 256, 256);
  CVT(lin2_W, w_l2,  512, 256, 256);
  CVT(head_W1, w_h1, 64, 512, 512);

  // ---- x = leaky(x @ lin1^T + b) ----
  CVT(x_in, xinp, N, 39, 64);
  launch_gemm(stream, xinp, 64, w_lin1, lin1_b, xA_f, xA_b, N, 256, ACT_LEAKY);
  float* xc_f = xA_f; unsigned short* xc_b = xA_b;
  float* xn_f = xB_f; unsigned short* xn_b = xB_b;

  int rd_blocks_N = (int)(((long)N * 32 + 127) / 128);
  int rd_blocks_G = (int)(((long)Gn * 32 + 127) / 128);

  // ---- GATEConv + gru0 ----
  launch_gemm(stream, xc_b, 256, w_g2, nullptr, hx_f, nullptr, N, 256, ACT_NONE); // xm = x@lin2^T
  k_rowdot<<<rd_blocks_N, 128, 0, stream>>>(xc_f, 256, g_attr, nullptr, xr, nullptr, N);
  FILL(akey, 0u, N); FILL(ssum, 0u, N); FILL(h_f, 0u, (long)N * 256);
  k_gate<<<E / 16, 128, 0, stream>>>(xc_b, eattr, src, dst, w_g1, g_attl, xr, alpha, akey);
  k_expsum<<<gs_blocks(E, 256), 256, 0, stream>>>(alpha, akey, dst, ebuf, ssum, E);
  k_scatter_msg<<<gs_blocks((long)E * 64, 256), 256, 0, stream>>>(ebuf, ssum, src, dst, hx_f, h_f, E);
  k_bias_act<<<gs_blocks((long)N * 256, 256), 256, 0, stream>>>(h_f, g_bias, h_b, 256, (long)N * 256, ACT_ELU);
  k_gru<<<N / 16, 128, 0, stream>>>(h_b, xc_b, xc_f, w_g0ih, w_g0hh, g0_bih, g0_bhh, xn_f, xn_b, N);
  { float* tf = xc_f; xc_f = xn_f; xn_f = tf; unsigned short* tb = xc_b; xc_b = xn_b; xn_b = tb; }

  // ---- atom GAT layers + GRUs ----
  for (int l = 0; l < 2; ++l) {
    launch_gemm(stream, xc_b, 256, w_aW[l], nullptr, hx_f, nullptr, N, 256, ACT_NONE);
    k_rowdot<<<rd_blocks_N, 128, 0, stream>>>(hx_f, 256, ac_asrc[l], ac_adst[l], asN, adN, N);
    FILL(akey, 0u, N); FILL(ssum, 0u, N); FILL(h_f, 0u, (long)N * 256);
    k_edge_logit<<<gs_blocks(E, 256), 256, 0, stream>>>(asN, adN, src, dst, alpha, akey, E);
    k_expsum<<<gs_blocks(E, 256), 256, 0, stream>>>(alpha, akey, dst, ebuf, ssum, E);
    k_scatter_msg<<<gs_blocks((long)E * 64, 256), 256, 0, stream>>>(ebuf, ssum, src, dst, hx_f, h_f, E);
    k_bias_act<<<gs_blocks((long)N * 256, 256), 256, 0, stream>>>(h_f, ac_b[l], h_b, 256, (long)N * 256, ACT_ELU);
    k_gru<<<N / 16, 128, 0, stream>>>(h_b, xc_b, xc_f, w_aih[l], w_ahh[l], ag_bih[l], ag_bhh[l], xn_f, xn_b, N);
    { float* tf = xc_f; xc_f = xn_f; xn_f = tf; unsigned short* tb = xc_b; xc_b = xn_b; xn_b = tb; }
  }

  // ---- molecule readout: out0 = relu(segment_sum(x, batch)) ----
  float* oc_f = oA_f; unsigned short* oc_b = oA_b;
  float* on_f = oB_f; unsigned short* on_b = oB_b;
  FILL(oc_f, 0u, (long)Gn * 256);
  k_scatter_rows<<<gs_blocks((long)N * 64, 256), 256, 0, stream>>>(xc_f, batch, oc_f, N);
  k_bias_act<<<gs_blocks((long)Gn * 256, 256), 256, 0, stream>>>(oc_f, nullptr, oc_b, 256, (long)Gn * 256, ACT_RELU);

  for (int t = 0; t < 2; ++t) {
    launch_gemm(stream, xc_b, 256, w_mW, nullptr, hx_f, nullptr, N, 256, ACT_NONE);  // hs
    launch_gemm(stream, oc_b, 256, w_mW, nullptr, hd_f, nullptr, Gn, 256, ACT_NONE); // hd
    k_rowdot<<<rd_blocks_N, 128, 0, stream>>>(hx_f, 256, mc_asrc, nullptr, asN, nullptr, N);
    k_rowdot<<<rd_blocks_G, 128, 0, stream>>>(hd_f, 256, mc_adst, nullptr, adG, nullptr, Gn);
    FILL(akg, 0u, Gn); FILL(ssg, 0u, Gn); FILL(mh_f, 0u, (long)Gn * 256);
    k_edge_logit<<<gs_blocks(N, 256), 256, 0, stream>>>(asN, adG, nullptr, batch, alpha, akg, N);
    k_expsum<<<gs_blocks(N, 256), 256, 0, stream>>>(alpha, akg, batch, ebuf, ssg, N);
    k_scatter_msg<<<gs_blocks((long)N * 64, 256), 256, 0, stream>>>(ebuf, ssg, nullptr, batch, hx_f, mh_f, N);
    k_bias_act<<<gs_blocks((long)Gn * 256, 256), 256, 0, stream>>>(mh_f, mc_b, mh_b, 256, (long)Gn * 256, ACT_ELU);
    k_gru<<<Gn / 16, 128, 0, stream>>>(mh_b, oc_b, oc_f, w_mih, w_mhh, mg_bih, mg_bhh, on_f, on_b, Gn);
    { float* tf = oc_f; oc_f = on_f; on_f = tf; unsigned short* tb = oc_b; oc_b = on_b; on_b = tb; }
  }

  // ---- output head ----
  launch_gemm(stream, oc_b, 256, w_l2, lin2_b, emb_f, emb_b, Gn, 512, ACT_NONE);
  launch_gemm(stream, emb_b, 512, w_h1, head_b1, h1_f, nullptr, Gn, 64, ACT_RELU);
  k_head2<<<Gn / 256, 256, 0, stream>>>(h1_f, head_W2, head_b2, (float*)d_out, Gn);

#undef CVT
#undef FILL
}
